// MoELayer_43224550867138
// MI455X (gfx1250) — matrix operand order repaired and checked by reading, not compiled, not run on hardware
//
#include <hip/hip_runtime.h>
#include <hip/hip_bf16.h>

// ---- problem constants (match reference) ----
#define DIM_  1024
#define HID_  2048
#define NE_   8
#define NTOK_ 8192   // B*N = 2*4096
#define TOPK_ 2

// ---- GEMM tiling ----
#define BM 128
#define BN 128
#define BK 32

typedef __attribute__((ext_vector_type(16))) __bf16 v16bf;
typedef __attribute__((ext_vector_type(4)))  __bf16 v4bf;
typedef __attribute__((ext_vector_type(8)))  float  v8f;

// ------------------------------------------------------------------
// WMMA wrapper: D = A(16x32 bf16) * B(32x16 bf16) + C(16x16 f32)
// ------------------------------------------------------------------
__device__ __forceinline__ v8f wmma_bf16(v16bf a, v16bf b, v8f c) {
    return __builtin_amdgcn_wmma_f32_16x16x32_bf16(
        /*neg_a=*/false, a, /*neg_b=*/false, b,
        /*c_mod=*/(short)0, c, /*reuse_a=*/false, /*reuse_b=*/false);
}

// A fragment (16x32, bf16) per ISA layout:
//  lanes 0-15 : row M=lane,    kb=0 : K = {0..7, 16..23}
//  lanes 16-31: row M=lane-16, kb=8 : K = {8..15, 24..31}
__device__ __forceinline__ v16bf load_frag_a(const __bf16* As /*[BM][BK]*/,
                                             int rowBase, int lane) {
    const int m  = rowBase + (lane & 15);
    const int kb = (lane < 16) ? 0 : 8;
    const __bf16* p = As + m * BK + kb;
    union { v16bf v; uint4 u[2]; } r;
    r.u[0] = *(const uint4*)(p);        // K = kb .. kb+7
    r.u[1] = *(const uint4*)(p + 16);   // K = kb+16 .. kb+23
    return r.v;
}

// B fragment (32x16, bf16): column N = lane&15,
//  lanes 0-15 hold K={0..7,16..23}, lanes 16-31 hold K={8..15,24..31}
__device__ __forceinline__ v16bf load_frag_b(const __bf16* Bs /*[BN][BK]*/,
                                             int colBase, int lane) {
    const int n  = colBase + (lane & 15);
    const int kb = (lane < 16) ? 0 : 8;
    const __bf16* p = Bs + n * BK + kb;
    union { v16bf v; uint4 u[2]; } r;
    r.u[0] = *(const uint4*)(p);
    r.u[1] = *(const uint4*)(p + 16);
    return r.v;
}

// ---- register-staged tile buffers (global -> regs -> LDS pipeline) ----
struct StagedF32 { float4 v[4]; };   // 128x32 f32 tile, 256 threads
struct StagedB16 { uint4  v[2]; };   // 128x32 bf16 tile, 256 threads

__device__ __forceinline__ void fetch_f32(StagedF32& s, const float* __restrict__ src,
                                          int row0, int k0, int ld, int tid) {
    const int c = (tid & 7) * 4;   // 8 threads * float4 = 32 cols
    const int r = tid >> 3;        // 32 rows per pass, 4 passes
#pragma unroll
    for (int i = 0; i < 4; ++i)
        s.v[i] = *(const float4*)(src + (size_t)(row0 + r + i * 32) * ld + k0 + c);
}

__device__ __forceinline__ void commit_f32(__bf16* dst /*[BM][BK]*/,
                                           const StagedF32& s, int tid) {
    const int c = (tid & 7) * 4;
    const int r = tid >> 3;
#pragma unroll
    for (int i = 0; i < 4; ++i) {
        v4bf b;
        b[0] = (__bf16)s.v[i].x; b[1] = (__bf16)s.v[i].y;
        b[2] = (__bf16)s.v[i].z; b[3] = (__bf16)s.v[i].w;
        *(v4bf*)(dst + (r + i * 32) * BK + c) = b;
    }
}

__device__ __forceinline__ void fetch_b16(StagedB16& s, const __bf16* __restrict__ src,
                                          int row0, int k0, int ld, int tid) {
    const int c = (tid & 3) * 8;   // 4 threads * 8 bf16 = 32 cols
    const int r = tid >> 2;        // 64 rows per pass, 2 passes
#pragma unroll
    for (int i = 0; i < 2; ++i)
        s.v[i] = *(const uint4*)(src + (size_t)(row0 + r + i * 64) * ld + k0 + c);
}

__device__ __forceinline__ void commit_b16(__bf16* dst /*[BM][BK]*/,
                                           const StagedB16& s, int tid) {
    const int c = (tid & 3) * 8;
    const int r = tid >> 2;
#pragma unroll
    for (int i = 0; i < 2; ++i)
        *(uint4*)(dst + (r + i * 64) * BK + c) = s.v[i];
}

// ------------------------------------------------------------------
// Router: logits = x @ router_w^T ; top-2 ; softmax over the 2.
// One wave per token (wave32), 8 waves per block.
// ------------------------------------------------------------------
__global__ __launch_bounds__(256) void moe_router_kernel(
    const float* __restrict__ x, const float* __restrict__ rw,
    float* __restrict__ wdense, float* __restrict__ rweights,
    int* __restrict__ sel) {
    const int lane = threadIdx.x & 31;
    const int wave = threadIdx.x >> 5;
    const int t = blockIdx.x * 8 + wave;

    const float* xr = x + (size_t)t * DIM_;
    float acc[NE_];
#pragma unroll
    for (int e = 0; e < NE_; ++e) acc[e] = 0.f;

    for (int d = lane; d < DIM_; d += 32) {
        const float xv = xr[d];
#pragma unroll
        for (int e = 0; e < NE_; ++e) acc[e] += xv * rw[e * DIM_ + d];
    }
#pragma unroll
    for (int e = 0; e < NE_; ++e)
#pragma unroll
        for (int off = 16; off > 0; off >>= 1)
            acc[e] += __shfl_xor(acc[e], off, 32);

    if (lane == 0) {
        float l0 = -3.4e38f, l1 = -3.4e38f;
        int i0 = 0, i1 = 0;
#pragma unroll
        for (int e = 0; e < NE_; ++e) {
            const float v = acc[e];
            if (v > l0)      { l1 = l0; i1 = i0; l0 = v; i0 = e; }
            else if (v > l1) { l1 = v; i1 = e; }
        }
        const float tt  = __expf(l1 - l0);              // l1 <= l0, stable
        const float inv = __builtin_amdgcn_rcpf(1.f + tt);
        const float w0 = inv, w1 = tt * inv;

        float wd[NE_];
#pragma unroll
        for (int e = 0; e < NE_; ++e) wd[e] = 0.f;
        wd[i0] = w0; wd[i1] = w1;
#pragma unroll
        for (int e = 0; e < NE_; ++e) wdense[(size_t)t * NE_ + e] = wd[e];

        rweights[(size_t)t * TOPK_ + 0] = w0;
        rweights[(size_t)t * TOPK_ + 1] = w1;
        sel[(size_t)t * TOPK_ + 0] = i0;
        sel[(size_t)t * TOPK_ + 1] = i1;
    }
}

// ------------------------------------------------------------------
// Stage 1 (per expert): H = silu(X @ W1^T) * (X @ W3^T), H stored bf16.
// NT GEMM, 128x128x32 tile, 8 waves (4x2), 2x4 frags/wave, dual B.
// Register double buffer for global loads + ping-pong LDS buffers:
// exactly one workgroup barrier per K-step.
// ------------------------------------------------------------------
__global__ __launch_bounds__(256) void moe_stage1_kernel(
    const float* __restrict__ x, const float* __restrict__ w1,
    const float* __restrict__ w3, __bf16* __restrict__ H, int e) {
    __shared__ __bf16 As[2][BM * BK];
    __shared__ __bf16 B1s[2][BN * BK];
    __shared__ __bf16 B3s[2][BN * BK];

    const int tid  = threadIdx.x;
    const int lane = tid & 31;
    const int wave = tid >> 5;
    const int wm   = wave >> 1;  // 0..3
    const int wn   = wave & 1;   // 0..1
    const int row0 = blockIdx.y * BM;
    const int col0 = blockIdx.x * BN;

    const float* w1e = w1 + (size_t)e * HID_ * DIM_;
    const float* w3e = w3 + (size_t)e * HID_ * DIM_;

    v8f acc1[2][4] = {};
    v8f acc3[2][4] = {};

    // prologue: stage tile kt=0 into LDS buffer 0
    StagedF32 sx, s1, s3;
    fetch_f32(sx, x,   row0, 0, DIM_, tid);
    fetch_f32(s1, w1e, col0, 0, DIM_, tid);
    fetch_f32(s3, w3e, col0, 0, DIM_, tid);
    commit_f32(As[0], sx, tid);
    commit_f32(B1s[0], s1, tid);
    commit_f32(B3s[0], s3, tid);
    __syncthreads();

    int cur = 0;
    for (int kt = 0;;) {
        const bool hasNext = (kt + BK) < DIM_;
        if (hasNext) {                     // issue global loads early
            fetch_f32(sx, x,   row0, kt + BK, DIM_, tid);
            fetch_f32(s1, w1e, col0, kt + BK, DIM_, tid);
            fetch_f32(s3, w3e, col0, kt + BK, DIM_, tid);
        }

        // load ALL fragments into distinct registers, then the WMMA block
        const v16bf a0 = load_frag_a(As[cur], wm * 32 + 0,  lane);
        const v16bf a1 = load_frag_a(As[cur], wm * 32 + 16, lane);
        v16bf bf1[4], bf3[4];
#pragma unroll
        for (int fn = 0; fn < 4; ++fn) {
            bf1[fn] = load_frag_b(B1s[cur], wn * 64 + fn * 16, lane);
            bf3[fn] = load_frag_b(B3s[cur], wn * 64 + fn * 16, lane);
        }
#pragma unroll
        for (int fn = 0; fn < 4; ++fn) {
            acc1[0][fn] = wmma_bf16(a0, bf1[fn], acc1[0][fn]);
            acc1[1][fn] = wmma_bf16(a1, bf1[fn], acc1[1][fn]);
            acc3[0][fn] = wmma_bf16(a0, bf3[fn], acc3[0][fn]);
            acc3[1][fn] = wmma_bf16(a1, bf3[fn], acc3[1][fn]);
        }

        if (!hasNext) break;
        // write NEXT tile into the other LDS buffer: no barrier needed before,
        // previous-iteration readers of that buffer were fenced by last barrier
        commit_f32(As[cur ^ 1], sx, tid);  // waits on loadcnt, after compute
        commit_f32(B1s[cur ^ 1], s1, tid);
        commit_f32(B3s[cur ^ 1], s3, tid);
        __syncthreads();
        cur ^= 1;
        kt += BK;
    }

    // Epilogue: h = silu(g)*u. C-layout: VGPR i -> M = i (+8 if lane>=16), N = lane&15.
    const int mHalf = (lane >> 4) * 8;
    const int nLane = lane & 15;
#pragma unroll
    for (int fm = 0; fm < 2; ++fm)
#pragma unroll
        for (int fn = 0; fn < 4; ++fn)
#pragma unroll
            for (int i = 0; i < 8; ++i) {
                const int m = row0 + wm * 32 + fm * 16 + mHalf + i;
                const int n = col0 + wn * 64 + fn * 16 + nLane;
                const float g = acc1[fm][fn][i];
                const float u = acc3[fm][fn][i];
                const float h = g * __builtin_amdgcn_rcpf(1.f + __expf(-g)) * u;
                H[(size_t)m * HID_ + n] = (__bf16)h;
            }
}

// ------------------------------------------------------------------
// Stage 2 (per expert): out (+)= (H @ W2^T) * w_dense[:, e]
// H:[NTOK,HID] bf16, W2:[DIM,HID] f32. Same pipelined structure.
// ------------------------------------------------------------------
__global__ __launch_bounds__(256) void moe_stage2_kernel(
    const __bf16* __restrict__ H, const float* __restrict__ w2,
    const float* __restrict__ wdense, float* __restrict__ out,
    int e, int accumulate) {
    __shared__ __bf16 As[2][BM * BK];
    __shared__ __bf16 Bs[2][BN * BK];
    __shared__ float  wdS[BM];

    const int tid  = threadIdx.x;
    const int lane = tid & 31;
    const int wave = tid >> 5;
    const int wm   = wave >> 1;
    const int wn   = wave & 1;
    const int row0 = blockIdx.y * BM;
    const int col0 = blockIdx.x * BN;

    const float* w2e = w2 + (size_t)e * DIM_ * HID_;

    if (tid < BM) wdS[tid] = wdense[(size_t)(row0 + tid) * NE_ + e];

    v8f acc[2][4] = {};

    StagedB16 sa;
    StagedF32 sb;
    fetch_b16(sa, H,   row0, 0, HID_, tid);
    fetch_f32(sb, w2e, col0, 0, HID_, tid);
    commit_b16(As[0], sa, tid);
    commit_f32(Bs[0], sb, tid);
    __syncthreads();

    int cur = 0;
    for (int kt = 0;;) {
        const bool hasNext = (kt + BK) < HID_;
        if (hasNext) {
            fetch_b16(sa, H,   row0, kt + BK, HID_, tid);
            fetch_f32(sb, w2e, col0, kt + BK, HID_, tid);
        }

        const v16bf a0 = load_frag_a(As[cur], wm * 32 + 0,  lane);
        const v16bf a1 = load_frag_a(As[cur], wm * 32 + 16, lane);
        v16bf bf[4];
#pragma unroll
        for (int fn = 0; fn < 4; ++fn)
            bf[fn] = load_frag_b(Bs[cur], wn * 64 + fn * 16, lane);
#pragma unroll
        for (int fn = 0; fn < 4; ++fn) {
            acc[0][fn] = wmma_bf16(a0, bf[fn], acc[0][fn]);
            acc[1][fn] = wmma_bf16(a1, bf[fn], acc[1][fn]);
        }

        if (!hasNext) break;
        commit_b16(As[cur ^ 1], sa, tid);
        commit_f32(Bs[cur ^ 1], sb, tid);
        __syncthreads();
        cur ^= 1;
        kt += BK;
    }

    const int mHalf = (lane >> 4) * 8;
    const int nLane = lane & 15;
#pragma unroll
    for (int fm = 0; fm < 2; ++fm)
#pragma unroll
        for (int fn = 0; fn < 4; ++fn)
#pragma unroll
            for (int i = 0; i < 8; ++i) {
                const int ml = wm * 32 + fm * 16 + mHalf + i;
                const int m  = row0 + ml;
                const int n  = col0 + wn * 64 + fn * 16 + nLane;
                const float v = acc[fm][fn][i] * wdS[ml];
                const size_t idx = (size_t)m * DIM_ + n;
                if (accumulate) out[idx] += v;
                else            out[idx]  = v;
            }
}

// ------------------------------------------------------------------
// Host launcher
// ------------------------------------------------------------------
extern "C" void kernel_launch(void* const* d_in, const int* in_sizes, int n_in,
                              void* d_out, int out_size, void* d_ws, size_t ws_size,
                              hipStream_t stream) {
    const float* x        = (const float*)d_in[0];
    const float* router_w = (const float*)d_in[1];
    const float* w1       = (const float*)d_in[2];
    const float* w3       = (const float*)d_in[3];
    const float* w2       = (const float*)d_in[4];

    // d_out tuple layout: out [8192*1024] f32 | routing_weights [8192*2] f32 |
    //                     selected_experts [8192*2] i32
    float* out      = (float*)d_out;
    float* rweights = out + (size_t)NTOK_ * DIM_;
    int*   sel      = (int*)(rweights + (size_t)NTOK_ * TOPK_);

    // workspace: w_dense [8192*8] f32 (256 KB), then H [8192*2048] bf16 (32 MB)
    float*  wdense  = (float*)d_ws;
    size_t  wdBytes = ((size_t)NTOK_ * NE_ * sizeof(float) + 255) & ~(size_t)255;
    __bf16* H       = (__bf16*)((char*)d_ws + wdBytes);

    moe_router_kernel<<<NTOK_ / 8, 256, 0, stream>>>(x, router_w, wdense,
                                                     rweights, sel);

    const dim3 g1(HID_ / BN, NTOK_ / BM);   // (16, 64)
    const dim3 g2(DIM_ / BN, NTOK_ / BM);   // (8, 64)
    for (int e = 0; e < NE_; ++e) {
        moe_stage1_kernel<<<g1, 256, 0, stream>>>(x, w1, w3, H, e);
        moe_stage2_kernel<<<g2, 256, 0, stream>>>(H, w2, wdense, out, e,
                                                  e > 0 ? 1 : 0);
    }
}
